// ProsodyExtractor_73598559584394
// MI455X (gfx1250) — matrix operand order repaired and checked by reading, not compiled, not run on hardware
//
#include <hip/hip_runtime.h>

// ---------------- problem constants ----------------
#define BB    32
#define TT    4096
#define PP    512
#define WW    128
#define NMEL  80
#define CIN1  96      // NMEL padded to multiple of 32 for WMMA K-chunks
#define FILT  256
#define PDIM  64
#define KTAPS 9
#define CPAD  4       // (K-1)/2
#define EPS_LN 1e-5f

#define LDS3 __attribute__((address_space(3)))

typedef __attribute__((ext_vector_type(16))) _Float16 v16h;
typedef __attribute__((ext_vector_type(8)))  float    v8f;

struct U4 { unsigned int x, y, z, w; };
union Frag { v16h v; U4 q[2]; };
union AccU { v8f v; float f[8]; };

// ---------------- packing kernels ----------------

// f32 mels [B,T,80] -> f16 [B,T,96] (zero-padded channels)
__global__ void pack_mels(const float* __restrict__ mels, _Float16* __restrict__ out) {
    size_t i = (size_t)blockIdx.x * 256 + threadIdx.x;
    size_t total = (size_t)BB * TT * CIN1;
    if (i >= total) return;
    int c = (int)(i % CIN1);
    size_t bt = i / CIN1;
    float v = (c < NMEL) ? mels[bt * NMEL + c] : 0.0f;
    out[i] = (_Float16)v;
}

// conv weight [Cout][Cin][Kt] f32 -> [Cout][Kt*CinPad] f16 (K-major contiguous per row)
__global__ void pack_conv_w(const float* __restrict__ w, _Float16* __restrict__ wp,
                            int Cout, int Cin, int CinPad, int Kt) {
    int i = blockIdx.x * 256 + threadIdx.x;
    int total = Cout * Kt * CinPad;
    if (i >= total) return;
    int ci = i % CinPad;
    int t  = i / CinPad;
    int k  = t % Kt;
    int co = t / Kt;
    float v = (ci < Cin) ? w[((size_t)co * Cin + ci) * Kt + k] : 0.0f;
    wp[(size_t)co * (Kt * CinPad) + (size_t)k * CinPad + ci] = (_Float16)v;
}

// mlp weight [Cin][Cout] f32 -> transposed [Cout][Cin] f16
__global__ void pack_mlp_w(const float* __restrict__ w, _Float16* __restrict__ wp,
                           int Cin, int Cout) {
    int i = blockIdx.x * 256 + threadIdx.x;
    if (i >= Cin * Cout) return;
    int co = i % Cout;
    int ci = i / Cout;
    wp[(size_t)co * Cin + ci] = (_Float16)w[(size_t)ci * Cout + co];
}

// ---------------- WMMA implicit-GEMM conv / matmul ----------------
// in  : [nb][Trows][Cin]  f16
// wp  : [Cout][taps*Cin]  f16 (per-row contiguous K-dim)
// out : [nb][Trows][Cout] f16, fused bias + ReLU
// 256 threads = 8 waves. Each wave: 16 rows x (16*NACC) couts. Block tile =
// 32 rows x (64*NACC) couts. Activation staging: double-buffered LDS fed by
// the CDNA5 async-copy pipe (ASYNCcnt), overlapped with the WMMA stream.
// Epilogue: LDS transpose tile -> global_store_async_from_lds_b128 (coalesced).
template <int NACC>
__global__ __launch_bounds__(256, 2) void conv_wmma(
    const _Float16* __restrict__ in, const _Float16* __restrict__ wp,
    const float* __restrict__ bias, _Float16* __restrict__ out,
    int Trows, int Cin, int Cout, int taps, int pad) {
    __shared__ _Float16 ldsA[2][32][256];   // 2 x 16 KB ping-pong slabs

    const int tid    = threadIdx.x;
    const int wave   = tid >> 5;
    const int lane   = tid & 31;
    const int laneHi = lane >> 4;
    const int lan    = lane & 15;
    const int t0     = blockIdx.x * 32;
    const int b      = blockIdx.y;
    const int tm     = wave & 1;          // row sub-tile (0..1)
    const int tg     = wave >> 1;         // cout group (0..3)
    const int colBase = blockIdx.z * (64 * NACC) + tg * (16 * NACC);
    const int Kdim   = taps * Cin;

    AccU acc[NACC];
    const _Float16* __restrict__ wrow[NACC];
#pragma unroll
    for (int j = 0; j < NACC; ++j) {
        acc[j].v = (v8f){0.f, 0.f, 0.f, 0.f, 0.f, 0.f, 0.f, 0.f};
        wrow[j] = wp + (size_t)(colBase + j * 16 + lan) * Kdim;
    }

    const int vecPerRow = Cin >> 3;       // 16B chunks per row

    // Issue one tap's activation slab into LDS buffer `bi` via the async pipe.
    auto stage = [&](int tap, int bi) {
        for (int idx = tid; idx < 32 * vecPerRow; idx += 256) {
            int r    = idx / vecPerRow;
            int c8   = (idx % vecPerRow) * 8;
            int srcT = t0 + r - pad + tap;
            if (srcT >= 0 && srcT < Trows) {
                const _Float16* gp = in + ((size_t)b * Trows + srcT) * Cin + c8;
                LDS3 _Float16* lp = (LDS3 _Float16*)&ldsA[bi][r][c8];
                asm volatile("global_load_async_to_lds_b128 %0, %1, off"
                             :: "v"(lp), "v"(gp) : "memory");
            } else {
                *(U4*)(&ldsA[bi][r][c8]) = U4{0u, 0u, 0u, 0u};
            }
        }
    };

    stage(0, 0);   // prologue: kick off first slab

    for (int tap = 0; tap < taps; ++tap) {
        // Drain my async copies for this tap, then rendezvous: after the
        // barrier the current slab is complete and the other slab is idle.
        asm volatile("s_wait_asynccnt 0x0" ::: "memory");
        __syncthreads();
        if (tap + 1 < taps) {
            stage(tap + 1, (tap + 1) & 1);   // overlap next slab with compute
            __builtin_prefetch(wrow[0] + (size_t)(tap + 1) * Cin, 0, 0);
        }
        const int bi = tap & 1;

        for (int kc = 0; kc < Cin; kc += 32) {
            // A fragment, 16x32 f16 per ISA layout: lanes 0-15 hold M=lane,
            // halves 0..7 -> K = kc+laneHi*8+h ; halves 8..15 -> K = kc+16+laneHi*8+h
            Frag a;
            const int arow = tm * 16 + lan;
            a.q[0] = *(const U4*)(&ldsA[bi][arow][kc + laneHi * 8]);
            a.q[1] = *(const U4*)(&ldsA[bi][arow][kc + 16 + laneHi * 8]);
#pragma unroll
            for (int j = 0; j < NACC; ++j) {
                // B fragment, 32x16 f16: lane's column N = lan, K same as A (laneHi*8 + h ; 16 + laneHi*8 + h)
                Frag bf;
                const _Float16* wk = wrow[j] + (size_t)tap * Cin + kc + laneHi * 8;
                bf.q[0] = *(const U4*)(wk);
                bf.q[1] = *(const U4*)(wk + 16);
                acc[j].v = __builtin_amdgcn_wmma_f32_16x16x32_f16(
                    false, a.v, false, bf.v, (short)0, acc[j].v, false, false);
            }
        }
    }

    // ---- epilogue: bias+ReLU into an LDS tile, then coalesced async stores ----
    __syncthreads();                       // all compute done; slabs reusable
    auto& ldsO = ldsA[0];                  // 32 x 256 f16 staging tile
#pragma unroll
    for (int j = 0; j < NACC; ++j) {
        const int colIn = tg * (16 * NACC) + j * 16 + lan;  // column inside block tile
        const float bv = bias[colBase + j * 16 + lan];
#pragma unroll
        for (int r = 0; r < 8; ++r) {
            int row = tm * 16 + laneHi * 8 + r;
            float v = acc[j].f[r] + bv;
            v = v > 0.0f ? v : 0.0f;
            ldsO[row][colIn] = (_Float16)v;
        }
    }
    __syncthreads();                       // DScnt drained: LDS tile visible to async engine

    const int colsPB   = 64 * NACC;
    const int colBlk0  = blockIdx.z * colsPB;
    const int vecsPerR = colsPB >> 3;
    for (int idx = tid; idx < 32 * vecsPerR; idx += 256) {
        int r  = idx / vecsPerR;
        int c8 = (idx % vecsPerR) * 8;
        _Float16* gp = out + ((size_t)b * Trows + t0 + r) * Cout + colBlk0 + c8;
        LDS3 const _Float16* lp = (LDS3 const _Float16*)&ldsO[r][c8];
        asm volatile("global_store_async_from_lds_b128 %0, %1, off"
                     :: "v"(gp), "v"(lp) : "memory");
    }
    // async stores drain at the implicit S_WAIT_IDLE of s_endpgm
}

// ---------------- in-place LayerNorm over 256 channels (input already ReLU'd) ----------------
__global__ __launch_bounds__(256) void ln_inplace(_Float16* __restrict__ buf,
                                                  const float* __restrict__ g,
                                                  const float* __restrict__ bt) {
    __shared__ float smem[8];
    __shared__ float stat[2];
    const size_t row = blockIdx.x;
    const int c = threadIdx.x;
    const int wave = c >> 5, lane = c & 31;

    float x = (float)buf[row * FILT + c];

    float s = x;
    for (int o = 16; o > 0; o >>= 1) s += __shfl_down(s, o);
    if (lane == 0) smem[wave] = s;
    __syncthreads();
    if (c == 0) { float t = 0.f; for (int i = 0; i < 8; ++i) t += smem[i]; stat[0] = t * (1.0f / FILT); }
    __syncthreads();
    const float m = stat[0];
    const float d = x - m;
    float s2 = d * d;
    for (int o = 16; o > 0; o >>= 1) s2 += __shfl_down(s2, o);
    if (lane == 0) smem[wave] = s2;
    __syncthreads();
    if (c == 0) { float t = 0.f; for (int i = 0; i < 8; ++i) t += smem[i]; stat[1] = t * (1.0f / FILT); }
    __syncthreads();
    const float y = d * rsqrtf(stat[1] + EPS_LN) * g[c] + bt[c];
    buf[row * FILT + c] = (_Float16)y;
}

// ---------------- segment maps (cumsums + phone->word ids) ----------------
__global__ void build_maps(const int* __restrict__ dur, const int* __restrict__ wlen,
                           int* __restrict__ phoneStart, int* __restrict__ wordStart,
                           int* __restrict__ pw) {
    int b = blockIdx.x * blockDim.x + threadIdx.x;
    if (b >= BB) return;
    int acc = 0;
    phoneStart[b * (PP + 1)] = 0;
    for (int p = 0; p < PP; ++p) { acc += dur[b * PP + p]; phoneStart[b * (PP + 1) + p + 1] = acc; }
    acc = 0;
    wordStart[b * (WW + 1)] = 0;
    for (int w = 0; w < WW; ++w) { acc += wlen[b * WW + w]; wordStart[b * (WW + 1) + w + 1] = acc; }
    int w = 0;
    for (int p = 0; p < PP; ++p) {
        while (w < WW && wordStart[b * (WW + 1) + w + 1] <= p) ++w;
        int wid = w; if (wid > WW - 1) wid = WW - 1;
        pw[b * PP + p] = wid;
    }
}

// ---------------- pooling ----------------
__global__ __launch_bounds__(256) void phone_pool(const _Float16* __restrict__ act,
                                                  const int* __restrict__ phoneStart,
                                                  const int* __restrict__ dur,
                                                  float* __restrict__ pm) {
    const int b = blockIdx.y, p = blockIdx.x, c = threadIdx.x;
    int s = phoneStart[b * (PP + 1) + p];
    int e = phoneStart[b * (PP + 1) + p + 1];
    if (s > TT) s = TT;
    if (e > TT) e = TT;
    float sum = 0.f;
    for (int t = s; t < e; ++t) sum += (float)act[((size_t)b * TT + t) * FILT + c];
    int d = dur[b * PP + p]; if (d < 1) d = 1;
    pm[((size_t)b * PP + p) * FILT + c] = sum / (float)d;
}

__global__ __launch_bounds__(256) void word_pool(const float* __restrict__ pm,
                                                 const int* __restrict__ wordStart,
                                                 const int* __restrict__ wlen,
                                                 _Float16* __restrict__ wm) {
    const int b = blockIdx.y, w = blockIdx.x, c = threadIdx.x;
    int s = wordStart[b * (WW + 1) + w];
    int e = wordStart[b * (WW + 1) + w + 1];
    if (s > PP) s = PP;
    if (e > PP) e = PP;
    float sum = 0.f;
    for (int p = s; p < e; ++p) sum += pm[((size_t)b * PP + p) * FILT + c];
    int d = wlen[b * WW + w]; if (d < 1) d = 1;
    wm[((size_t)b * WW + w) * FILT + c] = (_Float16)(sum / (float)d);
}

// ---------------- expand word emb back to phones + mask ----------------
__global__ void expand_mask(const _Float16* __restrict__ wemb, const int* __restrict__ pw,
                            const unsigned char* __restrict__ mask, float* __restrict__ out) {
    int i = blockIdx.x * 256 + threadIdx.x;   // over B*P*PDIM
    if (i >= BB * PP * PDIM) return;
    int d = i % PDIM;
    int bp = i / PDIM;
    int b = bp / PP;
    int wid = pw[bp];
    float v = mask[bp] ? 0.0f : (float)wemb[((size_t)b * WW + wid) * PDIM + d];
    out[i] = v;
}

// ---------------- host-side launch ----------------
extern "C" void kernel_launch(void* const* d_in, const int* in_sizes, int n_in,
                              void* d_out, int out_size, void* d_ws, size_t ws_size,
                              hipStream_t stream) {
    const unsigned char* mask = (const unsigned char*)d_in[0];
    const float* mels = (const float*)d_in[1];
    /* d_in[2] = mel_mask, unused by the reference */
    const int* durations = (const int*)d_in[3];
    const int* wlen      = (const int*)d_in[4];
    const float* c1w = (const float*)d_in[5];
    const float* c1b = (const float*)d_in[6];
    const float* ln1g = (const float*)d_in[7];
    const float* ln1b = (const float*)d_in[8];
    const float* c2w = (const float*)d_in[9];
    const float* c2b = (const float*)d_in[10];
    const float* ln2g = (const float*)d_in[11];
    const float* ln2b = (const float*)d_in[12];
    const float* w1 = (const float*)d_in[13];
    const float* b1 = (const float*)d_in[14];
    const float* w2 = (const float*)d_in[15];
    const float* b2 = (const float*)d_in[16];
    float* out = (float*)d_out;

    // bump allocator over workspace
    char* ws = (char*)d_ws;
    auto alloc = [&](size_t bytes) -> char* {
        char* p = ws;
        ws += (bytes + 255) & ~(size_t)255;
        return p;
    };
    _Float16* mels16 = (_Float16*)alloc((size_t)BB * TT * CIN1 * 2);         // 24 MB
    _Float16* wp1    = (_Float16*)alloc((size_t)FILT * KTAPS * CIN1 * 2);    // conv1 weights
    _Float16* wp2    = (_Float16*)alloc((size_t)FILT * KTAPS * FILT * 2);    // conv2 weights
    _Float16* wpm1   = (_Float16*)alloc((size_t)FILT * FILT * 2);            // w1^T
    _Float16* wpm2   = (_Float16*)alloc((size_t)PDIM * FILT * 2);            // w2^T
    _Float16* actA   = (_Float16*)alloc((size_t)BB * TT * FILT * 2);         // 64 MB
    _Float16* actB   = (_Float16*)alloc((size_t)BB * TT * FILT * 2);         // 64 MB
    float*    pm     = (float*)alloc((size_t)BB * PP * FILT * 4);            // phone means
    _Float16* wm     = (_Float16*)alloc((size_t)BB * WW * FILT * 2);         // word means (f16)
    _Float16* h1     = (_Float16*)alloc((size_t)BB * WW * FILT * 2);         // mlp1 out
    _Float16* wemb   = (_Float16*)alloc((size_t)BB * WW * PDIM * 2);         // mlp2 out
    int* phoneStart  = (int*)alloc((size_t)BB * (PP + 1) * 4);
    int* wordStart   = (int*)alloc((size_t)BB * (WW + 1) * 4);
    int* pw          = (int*)alloc((size_t)BB * PP * 4);

    // 1) packing
    {
        size_t tot = (size_t)BB * TT * CIN1;
        pack_mels<<<(unsigned)((tot + 255) / 256), 256, 0, stream>>>(mels, mels16);
    }
    pack_conv_w<<<(FILT * KTAPS * CIN1 + 255) / 256, 256, 0, stream>>>(c1w, wp1, FILT, NMEL, CIN1, KTAPS);
    pack_conv_w<<<(FILT * KTAPS * FILT + 255) / 256, 256, 0, stream>>>(c2w, wp2, FILT, FILT, FILT, KTAPS);
    pack_mlp_w<<<(FILT * FILT + 255) / 256, 256, 0, stream>>>(w1, wpm1, FILT, FILT);
    pack_mlp_w<<<(FILT * PDIM + 255) / 256, 256, 0, stream>>>(w2, wpm2, FILT, PDIM);
    build_maps<<<1, 32, 0, stream>>>(durations, wlen, phoneStart, wordStart, pw);

    // 2) conv1 (Kdim = 9*96 = 864) + LN   [block tile 32 x 256 -> gridDim.z = 1]
    conv_wmma<4><<<dim3(TT / 32, BB, 1), 256, 0, stream>>>(
        mels16, wp1, c1b, actA, TT, CIN1, FILT, KTAPS, CPAD);
    ln_inplace<<<BB * TT, 256, 0, stream>>>(actA, ln1g, ln1b);

    // 3) conv2 (Kdim = 9*256 = 2304) + LN
    conv_wmma<4><<<dim3(TT / 32, BB, 1), 256, 0, stream>>>(
        actA, wp2, c2b, actB, TT, FILT, FILT, KTAPS, CPAD);
    ln_inplace<<<BB * TT, 256, 0, stream>>>(actB, ln2g, ln2b);

    // 4) pooling: frames -> phones -> words
    phone_pool<<<dim3(PP, BB), 256, 0, stream>>>(actB, phoneStart, durations, pm);
    word_pool<<<dim3(WW, BB), 256, 0, stream>>>(pm, wordStart, wlen, wm);

    // 5) MLPs via the same WMMA GEMM kernel (taps=1, pad=0, rows = B*W = 4096)
    conv_wmma<4><<<dim3((BB * WW) / 32, 1, 1), 256, 0, stream>>>(
        wm, wpm1, b1, h1, BB * WW, FILT, FILT, 1, 0);
    conv_wmma<1><<<dim3((BB * WW) / 32, 1, 1), 256, 0, stream>>>(
        h1, wpm2, b2, wemb, BB * WW, FILT, PDIM, 1, 0);

    // 6) expand to phones + mask
    expand_mask<<<(BB * PP * PDIM + 255) / 256, 256, 0, stream>>>(wemb, pw, mask, out);
}
